// Dynamics_35210141893057
// MI455X (gfx1250) — matrix operand order repaired and checked by reading, not compiled, not run on hardware
//
#include <hip/hip_runtime.h>

// ---- problem constants ----
#define BATCH 128
#define TLEN  256
#define ADIM  8
#define HDIM  1024
#define H2    2048
#define H3    3072

// ---- vector types (plain ext_vector, safe in unions) ----
typedef __attribute__((ext_vector_type(16))) __bf16 bf16x16;
typedef __attribute__((ext_vector_type(8)))  float  f32x8;
typedef __attribute__((ext_vector_type(4)))  unsigned int u32x4;
typedef __attribute__((ext_vector_type(4)))  float  f32x4;
typedef unsigned short ushort_t;

// f32 -> bf16 (round to nearest even)
__device__ __forceinline__ ushort_t f2bf(float f) {
    unsigned int u = __float_as_uint(f);
    u += 0x7FFFu + ((u >> 16) & 1u);
    return (ushort_t)(u >> 16);
}

// A-matrix fragment, 16x32 bf16 (ISA 7.12.2 "16-bit A-Matrix 16x32"):
// lanes 0-15 (row = lane): v0-3 K=k0..k0+7,  v4-7 K=k0+16..k0+23
// lanes 16-31 (row = lane-16): v0-3 K=k0+8..k0+15, v4-7 K=k0+24..k0+31
__device__ __forceinline__ bf16x16 load_frag_a(const ushort_t* __restrict__ base,
                                               long ld, int row, int k0, int half) {
    const ushort_t* p = base + (long)row * ld + k0 + half * 8;
    union { u32x4 u[2]; bf16x16 v; } x;
    x.u[0] = *(const u32x4*)(p);
    x.u[1] = *(const u32x4*)(p + 16);
    return x.v;
}

// B-matrix fragment, 32x16 bf16 (per ISA 7.12.4 dense-B pattern scaled to K=32):
// lanes 0-15 (col = lane): v0-3 K=k0..k0+7,  v4-7 K=k0+16..k0+23
// lanes 16-31 (col = lane-16): v0-3 K=k0+8..k0+15, v4-7 K=k0+24..k0+31
// (same K order as the A fragment). Here "row" is the N index and the
// weight matrix is stored [N, K] row-major (w_ih/w_hh are [3H, H]), so K is
// contiguous: two b128 loads 16 elements apart.
__device__ __forceinline__ bf16x16 load_frag_b(const ushort_t* __restrict__ base,
                                               long ld, int col, int k0, int half) {
    const ushort_t* p = base + (long)col * ld + k0 + half * 8;
    union { u32x4 u[2]; bf16x16 v; } x;
    x.u[0] = *(const u32x4*)(p);
    x.u[1] = *(const u32x4*)(p + 16);
    return x.v;
}

// ---- prologue kernels ----
__global__ void convert_bf16_kernel(const float* __restrict__ in,
                                    ushort_t* __restrict__ out, int n) {
    for (int i = blockIdx.x * blockDim.x + threadIdx.x; i < n;
         i += gridDim.x * blockDim.x)
        out[i] = f2bf(in[i]);
}

__global__ void init_h_kernel(const float* __restrict__ hs,
                              float* __restrict__ h_f32,
                              ushort_t* __restrict__ h_bf16) {
    int i = blockIdx.x * blockDim.x + threadIdx.x;   // B*H threads
    float v = hs[i];
    h_f32[i]  = v;
    h_bf16[i] = f2bf(v);
}

// x[b,t,h] = relu(sum_a actions[b,t,a]*fc_w[h,a] + fc_b[h]) -> bf16
__global__ void input_proj_kernel(const float* __restrict__ actions,
                                  const float* __restrict__ fc_w,
                                  const float* __restrict__ fc_b,
                                  ushort_t* __restrict__ x_bf16) {
    long i  = (long)blockIdx.x * blockDim.x + threadIdx.x;  // over B*T*H
    int  j  = (int)(i & (HDIM - 1));
    long bt = i >> 10;
    const f32x4* ap = (const f32x4*)(actions + bt * ADIM);
    const f32x4* wp = (const f32x4*)(fc_w + (long)j * ADIM);
    f32x4 a0 = ap[0], a1 = ap[1], w0 = wp[0], w1 = wp[1];
    float s = fc_b[j];
    s += a0.x * w0.x + a0.y * w0.y + a0.z * w0.z + a0.w * w0.w;
    s += a1.x * w1.x + a1.y * w1.y + a1.z * w1.z + a1.w * w1.w;
    x_bf16[i] = f2bf(fmaxf(s, 0.0f));
}

// ---- per-step WMMA kernel ----
// Each wave owns one 16x16 output tile; computes acc_x = x_t @ w_ih^T and
// acc_h = h @ w_hh^T over K=1024 (32 k-chunks, 2 WMMAs each). Writes
// pre_rz = acc_x + acc_h + biases for gate cols < 2H, and the n-gate halves
// (xn_pre / hn_pre) separately since n = tanh(xn + r*hn) needs them split.
__global__ __launch_bounds__(128)
void gru_gemm_kernel(const ushort_t* __restrict__ x_bf16,
                     const ushort_t* __restrict__ h_bf16,
                     const ushort_t* __restrict__ wih,
                     const ushort_t* __restrict__ whh,
                     const float* __restrict__ b_ih,
                     const float* __restrict__ b_hh,
                     float* __restrict__ pre_rz,
                     float* __restrict__ xn_pre,
                     float* __restrict__ hn_pre,
                     int t) {
    const int wave = blockIdx.x * 4 + (threadIdx.x >> 5);
    const int lane = threadIdx.x & 31;
    const int half = lane >> 4;
    const int rl   = lane & 15;

    const int NT = H3 / 16;                 // 192 N-tiles
    const int mt = wave / NT;               // 0..7
    const int nt = wave % NT;               // 0..191
    const int m0 = mt * 16, n0 = nt * 16;

    const ushort_t* xbase = x_bf16 + (long)t * HDIM;   // row stride T*H

    f32x8 accx = {0.f,0.f,0.f,0.f,0.f,0.f,0.f,0.f};
    f32x8 acch = {0.f,0.f,0.f,0.f,0.f,0.f,0.f,0.f};

    #pragma unroll 4
    for (int k0 = 0; k0 < HDIM; k0 += 32) {
        bf16x16 ax = load_frag_a(xbase,  (long)TLEN * HDIM, m0 + rl, k0, half);
        bf16x16 ah = load_frag_a(h_bf16, HDIM,              m0 + rl, k0, half);
        bf16x16 bi = load_frag_b(wih,    HDIM,              n0 + rl, k0, half);
        bf16x16 bh = load_frag_b(whh,    HDIM,              n0 + rl, k0, half);
        accx = __builtin_amdgcn_wmma_f32_16x16x32_bf16(
                   false, ax, false, bi, (short)0, accx, false, false);
        acch = __builtin_amdgcn_wmma_f32_16x16x32_bf16(
                   false, ah, false, bh, (short)0, acch, false, false);
    }

    const int n    = n0 + rl;
    const float bx  = b_ih[n];
    const float bh2 = b_hh[n];
    #pragma unroll
    for (int v = 0; v < 8; ++v) {
        const int m = m0 + v + half * 8;      // C layout: lanes 16-31 -> M+8
        const float ax = accx[v] + bx;
        const float ah = acch[v] + bh2;
        if (n < H2) {
            pre_rz[(long)m * H2 + n] = ax + ah;
        } else {
            xn_pre[(long)m * HDIM + (n - H2)] = ax;
            hn_pre[(long)m * HDIM + (n - H2)] = ah;
        }
    }
}

// ---- per-step elementwise GRU update ----
__global__ void gru_elem_kernel(const float* __restrict__ pre_rz,
                                const float* __restrict__ xn_pre,
                                const float* __restrict__ hn_pre,
                                float* __restrict__ h_f32,
                                ushort_t* __restrict__ h_bf16,
                                float* __restrict__ out,
                                int t) {
    int i = blockIdx.x * blockDim.x + threadIdx.x;   // over B*H
    int b = i >> 10;
    int j = i & (HDIM - 1);
    float r = 1.0f / (1.0f + __expf(-pre_rz[(long)b * H2 + j]));
    float z = 1.0f / (1.0f + __expf(-pre_rz[(long)b * H2 + HDIM + j]));
    float nv = tanhf(xn_pre[i] + r * hn_pre[i]);
    float h  = h_f32[i];
    float hn = (1.0f - z) * nv + z * h;
    h_f32[i]  = hn;
    h_bf16[i] = f2bf(hn);
    out[(long)b * TLEN * HDIM + (long)t * HDIM + j] = hn;
}

extern "C" void kernel_launch(void* const* d_in, const int* in_sizes, int n_in,
                              void* d_out, int out_size, void* d_ws, size_t ws_size,
                              hipStream_t stream) {
    const float* actions = (const float*)d_in[0];   // [B,T,A]
    const float* hidden  = (const float*)d_in[1];   // [B,H]
    const float* fc_w    = (const float*)d_in[2];   // [H,A]
    const float* fc_b    = (const float*)d_in[3];   // [H]
    const float* w_ih    = (const float*)d_in[4];   // [3H,H]
    const float* w_hh    = (const float*)d_in[5];   // [3H,H]
    const float* b_ih    = (const float*)d_in[6];   // [3H]
    const float* b_hh    = (const float*)d_in[7];   // [3H]
    float* out = (float*)d_out;                     // [B,T,H]

    // ---- workspace carve-up (all offsets 256B aligned) ----
    char* ws = (char*)d_ws;
    size_t off = 0;
    ushort_t* x_bf16 = (ushort_t*)(ws + off); off += (size_t)BATCH*TLEN*HDIM*2; // 64 MB
    ushort_t* wih_bf = (ushort_t*)(ws + off); off += (size_t)H3*HDIM*2;         // 6 MB
    ushort_t* whh_bf = (ushort_t*)(ws + off); off += (size_t)H3*HDIM*2;         // 6 MB
    ushort_t* h_bf16 = (ushort_t*)(ws + off); off += (size_t)BATCH*HDIM*2;
    float*    h_f32  = (float*)(ws + off);    off += (size_t)BATCH*HDIM*4;
    float*    pre_rz = (float*)(ws + off);    off += (size_t)BATCH*H2*4;
    float*    xn_pre = (float*)(ws + off);    off += (size_t)BATCH*HDIM*4;
    float*    hn_pre = (float*)(ws + off);    off += (size_t)BATCH*HDIM*4;
    (void)ws_size; (void)in_sizes; (void)n_in; (void)out_size;

    // ---- prologue ----
    convert_bf16_kernel<<<4096, 256, 0, stream>>>(w_ih, wih_bf, H3 * HDIM);
    convert_bf16_kernel<<<4096, 256, 0, stream>>>(w_hh, whh_bf, H3 * HDIM);
    init_h_kernel<<<(BATCH * HDIM) / 256, 256, 0, stream>>>(hidden, h_f32, h_bf16);
    input_proj_kernel<<<(BATCH * TLEN * HDIM) / 256, 256, 0, stream>>>(
        actions, fc_w, fc_b, x_bf16);

    // ---- sequential recurrence ----
    // tiles = (128/16) * (3072/16) = 1536 waves; 4 waves/block -> 384 blocks
    for (int t = 0; t < TLEN; ++t) {
        gru_gemm_kernel<<<384, 128, 0, stream>>>(
            x_bf16, h_bf16, wih_bf, whh_bf, b_ih, b_hh,
            pre_rz, xn_pre, hn_pre, t);
        gru_elem_kernel<<<(BATCH * HDIM) / 256, 256, 0, stream>>>(
            pre_rz, xn_pre, hn_pre, h_f32, h_bf16, out, t);
    }
}
